// TransformerEncoderLayer_64046552318363
// MI455X (gfx1250) — matrix operand order repaired and checked by reading, not compiled, not run on hardware
//
#include <hip/hip_runtime.h>
#include <cstdint>

// ---------------------------------------------------------------------------
// TransformerEncoderLayer for MI455X (gfx1250, wave32, WMMA bf16 16x16x32,
// async-to-LDS double-buffered staging, key-major V for b32 LDS gathers)
// ---------------------------------------------------------------------------
constexpr int B_  = 8;
constexpr int S_  = 2048;
constexpr int D_  = 256;
constexpr int DK_ = 32;
constexpr int ROWS = B_ * S_;

typedef __attribute__((ext_vector_type(16))) __bf16 bf16x16;
typedef __attribute__((ext_vector_type(8)))  float  f32x8;

union ABu { unsigned int u[8]; bf16x16 v; };

static __device__ __forceinline__ unsigned short f2bf(float f) {
  unsigned int u = __float_as_uint(f);
  unsigned int r = 0x7FFFu + ((u >> 16) & 1u);   // round-to-nearest-even
  return (unsigned short)((u + r) >> 16);
}

static __device__ __forceinline__ f32x8 wmma_bf16(const ABu& A, const ABu& B, f32x8 C) {
  return __builtin_amdgcn_wmma_f32_16x16x32_bf16(
      /*neg_a=*/false, A.v, /*neg_b=*/false, B.v,
      /*c_mod=*/(short)0, C, /*reuse_a=*/false, /*reuse_b=*/false);
}

// A-operand (16xK=32 bf16) VGPR layout: lane m = lane&15,
// vgpr i holds K pair starting at (i>>2)*16 + ((lane>>4)&1)*8 + (i&3)*2
static __device__ __forceinline__ int a_k0(int i, int lh) {
  return (i >> 2) * 16 + lh * 8 + (i & 3) * 2;
}

// Async global->LDS copies (ASYNCcnt). lds = 32-bit LDS byte offset.
static __device__ __forceinline__ void async_b128(unsigned lds, unsigned long long ga) {
  asm volatile("global_load_async_to_lds_b128 %0, %1, off" :: "v"(lds), "v"(ga) : "memory");
}
static __device__ __forceinline__ void async_b64(unsigned lds, unsigned long long ga) {
  asm volatile("global_load_async_to_lds_b64 %0, %1, off" :: "v"(lds), "v"(ga) : "memory");
}
static __device__ __forceinline__ unsigned lds_off(const void* p) {
  return (unsigned)(size_t)p;   // generic ptr low 32 bits = LDS offset
}

// ---------------------------------------------------------------------------
// Kernel 0: Wff fp32 -> bf16
// ---------------------------------------------------------------------------
__global__ void wconv_kernel(const float* __restrict__ w,
                             unsigned short* __restrict__ wb, int n) {
  int i = blockIdx.x * blockDim.x + threadIdx.x;
  if (i < n) wb[i] = f2bf(w[i]);
}

// ---------------------------------------------------------------------------
// Kernel 1: q = l2norm(x@Wq)/sqrt(dk), k = l2norm(x@Wk)  (bf16, row-major)
//           vbT = (x@Wv)^T per batch: [B][col][s]  (bf16, key-major)
// Thread tid owns V column tid for its 16 rows -> transposed store is free.
// ---------------------------------------------------------------------------
__global__ __launch_bounds__(256) void qkv_kernel(
    const float* __restrict__ x, const float* __restrict__ Wq,
    const float* __restrict__ Wk, const float* __restrict__ Wv,
    unsigned short* __restrict__ qb, unsigned short* __restrict__ kb,
    unsigned short* __restrict__ vbT) {
  __shared__ float x_s[16][D_];
  __shared__ float q_s[16][DK_];
  __shared__ float k_s[16][DK_];
  __shared__ float nrm[2][16];

  const int tid = threadIdx.x;
  const size_t row0 = (size_t)blockIdx.x * 16;

#pragma unroll
  for (int j = 0; j < 16; j++) {
    int idx = tid + j * 256;
    x_s[idx >> 8][idx & 255] = x[row0 * D_ + idx];
  }
  __syncthreads();

#pragma unroll
  for (int j = 0; j < 4; j++) {
    int o = tid + j * 256;
    int r = o >> 6, which = (o >> 5) & 1, n = o & 31;
    const float* W = which ? Wk : Wq;
    float acc = 0.f;
#pragma unroll 4
    for (int kk = 0; kk < D_; kk++) acc += x_s[r][kk] * W[kk * DK_ + n];
    if (which) k_s[r][n] = acc; else q_s[r][n] = acc;
  }

  // V column `tid` for 16 rows -> key-major store (two 16B chunks)
  {
    __align__(16) unsigned short vloc[16];
#pragma unroll
    for (int j = 0; j < 16; j++) {
      float acc = 0.f;
#pragma unroll 4
      for (int kk = 0; kk < D_; kk++) acc += x_s[j][kk] * Wv[kk * D_ + tid];
      vloc[j] = f2bf(acc);
    }
    const int b  = (int)(row0 / S_);
    const int s0 = (int)(row0 % S_);
    unsigned short* dst = vbT + ((size_t)b * D_ + tid) * S_ + s0;
    *(uint4*)(dst)     = *(const uint4*)&vloc[0];
    *(uint4*)(dst + 8) = *(const uint4*)&vloc[8];
  }
  __syncthreads();

  if (tid < 32) {
    int which = tid >> 4, r = tid & 15;
    const float* src = which ? k_s[r] : q_s[r];
    float s = 0.f;
#pragma unroll
    for (int n = 0; n < DK_; n++) { float t = src[n]; s += t * t; }
    nrm[which][r] = fmaxf(sqrtf(s), 1e-12f);
  }
  __syncthreads();

#pragma unroll
  for (int j = 0; j < 4; j++) {
    int o = tid + j * 256;
    int r = o >> 6, which = (o >> 5) & 1, n = o & 31;
    if (which)
      kb[(row0 + r) * DK_ + n] = f2bf(k_s[r][n] / nrm[1][r]);
    else  // fold 1/sqrt(32) score scale into q
      qb[(row0 + r) * DK_ + n] =
          f2bf(q_s[r][n] / nrm[0][r] * 0.17677669529663687f);
  }
}

// ---------------------------------------------------------------------------
// Kernel 2: fused streaming attention, async double-buffered K/V/mask tiles.
// grid (S/64, B), 256 threads = 8 waves. Wave w: q-tile (w&3), N-half (w>>2).
// Per 32-key tile: 2 score WMMAs (dup per pair) + 1 rowsum WMMA + 8 PV WMMAs.
// V arrives key-major -> both K and V B-operand gathers are paired b32 reads.
// ---------------------------------------------------------------------------
__global__ __launch_bounds__(256) void attn_kernel(
    const unsigned short* __restrict__ qb, const unsigned short* __restrict__ kb,
    const unsigned short* __restrict__ vbT, const unsigned char* __restrict__ mask,
    float* __restrict__ sa) {
  constexpr int BQ = 64, TK = 32;
  __shared__ unsigned short vT_s[2][D_][40];     // [col][key], 80B rows
  __shared__ unsigned short k_s[2][TK][40];      // [key][dk],  80B rows
  __shared__ unsigned char  m_s[2][BQ][TK];
  __shared__ unsigned short p_s[4][16][34];      // per q-tile P (C->A transpose)

  const int tid  = threadIdx.x;
  const int wave = tid >> 5, lane = tid & 31;
  const int qt = wave & 3, nh = wave >> 2;
  const int ln = lane & 15, lh = (lane >> 4) & 1;
  const int b  = blockIdx.y;
  const int q0 = blockIdx.x * BQ;

  // q tile A operand, resident all kernel
  ABu qA;
  {
    const unsigned short* qp = qb + ((size_t)b * S_ + q0 + qt * 16 + ln) * DK_;
#pragma unroll
    for (int i = 0; i < 8; i++)
      qA.u[i] = *(const unsigned int*)(qp + a_k0(i, lh));
  }

  f32x8 acc[8];
#pragma unroll
  for (int i = 0; i < 8; i++) acc[i] = f32x8{};
  f32x8 accsum{};
  ABu ones;
#pragma unroll
  for (int i = 0; i < 8; i++) ones.u[i] = 0x3F803F80u;  // bf16 1.0 pairs

  // Stage one 32-key tile: 1024 V + 128 K + 128 mask b128 chunks = 5/thread.
  // Region boundaries are multiples of 32 -> per-wave async count == 5 exactly.
  auto stage = [&](int t0, int buf) {
#pragma unroll
    for (int j = 0; j < 5; j++) {
      int c = tid + j * 256;
      unsigned lds; unsigned long long ga;
      if (c < 1024) {                 // V tile: 256 cols x 64B (key-major)
        int col = c >> 2, cho = (c & 3) * 8;
        lds = lds_off(&vT_s[buf][col][cho]);
        ga  = (unsigned long long)(vbT + ((size_t)b * D_ + col) * S_ + t0 + cho);
      } else if (c < 1152) {          // K tile: 32 keys x 64B
        int u = c - 1024; int key = u >> 2, pos = (u & 3) * 8;
        lds = lds_off(&k_s[buf][key][pos]);
        ga  = (unsigned long long)(kb + ((size_t)b * S_ + t0 + key) * DK_ + pos);
      } else {                        // mask tile: 64 rows x 32B
        int u = c - 1152; int r = u >> 1, o16 = (u & 1) * 16;
        lds = lds_off(&m_s[buf][r][o16]);
        ga  = (unsigned long long)(mask + ((size_t)b * S_ + q0 + r) * S_ + t0 + o16);
      }
      async_b128(lds, ga);
    }
  };

  stage(0, 0);
  for (int t0 = 0; t0 < S_; t0 += TK) {
    const int buf = (t0 / TK) & 1;
    if (t0 + TK < S_) {
      stage(t0 + TK, buf ^ 1);                           // prefetch next tile
      asm volatile("s_wait_asynccnt 0x5" ::: "memory");  // tile t landed
    } else {
      asm volatile("s_wait_asynccnt 0x0" ::: "memory");
    }
    __syncthreads();

    // scores, exp, mask -> P (bf16) ; duplicated across the nh pair
#pragma unroll
    for (int st = 0; st < 2; st++) {
      ABu kB;
#pragma unroll
      for (int j = 0; j < 8; j++)
        kB.u[j] = *(const unsigned int*)&k_s[buf][st * 16 + ln][a_k0(j, lh)];
      f32x8 sc = wmma_bf16(qA, kB, f32x8{});
      if (nh == 0) {
#pragma unroll
        for (int r = 0; r < 8; r++) {
          int m = r + lh * 8, keyc = st * 16 + ln;
          float z = m_s[buf][qt * 16 + m][keyc] ? 0.f : __expf(sc[r]);
          p_s[qt][m][keyc] = f2bf(z);
        }
      }
    }
    __syncthreads();  // P visible cross-lane

    ABu pA;
#pragma unroll
    for (int i = 0; i < 8; i++)
      pA.u[i] = *(const unsigned int*)&p_s[qt][ln][a_k0(i, lh)];

    // row sums via all-ones B: same C layout as acc -> no cross-lane shuffles
    accsum = wmma_bf16(pA, ones, accsum);
#pragma unroll
    for (int i = 0; i < 8; i++) {
      int nt = nh * 8 + i;
      ABu vB;
#pragma unroll
      for (int j = 0; j < 8; j++)
        vB.u[j] = *(const unsigned int*)&vT_s[buf][nt * 16 + ln][a_k0(j, lh)];
      acc[i] = wmma_bf16(pA, vB, acc[i]);
    }
    __syncthreads();  // all reads of buf done -> next stage may overwrite
  }

  // sa = acc / (rowsum + 1)
#pragma unroll
  for (int r = 0; r < 8; r++) {
    int m = r + lh * 8;
    size_t row = (size_t)b * S_ + q0 + qt * 16 + m;
    float denom = accsum[r] + 1.0f;
#pragma unroll
    for (int i = 0; i < 8; i++)
      sa[row * D_ + (nh * 8 + i) * 16 + ln] = acc[i][r] / denom;
  }
}

// ---------------------------------------------------------------------------
// Kernel 3: zz = LayerNorm(x + sa) -> zz fp32 + zzb bf16 (wave per row)
// ---------------------------------------------------------------------------
__global__ __launch_bounds__(256) void ln1_kernel(
    const float* __restrict__ x, const float* __restrict__ sa,
    const float* __restrict__ g1, const float* __restrict__ b1,
    float* __restrict__ zz, unsigned short* __restrict__ zzb) {
  const int tid = threadIdx.x;
  const int wave = tid >> 5, lane = tid & 31;
  size_t row = (size_t)blockIdx.x * 8 + wave;
  const float* xr = x + row * D_;
  const float* sr = sa + row * D_;
  float t[8], s = 0.f, s2 = 0.f;
#pragma unroll
  for (int i = 0; i < 8; i++) {
    int c = lane + i * 32;
    t[i] = xr[c] + sr[c];
    s += t[i]; s2 += t[i] * t[i];
  }
#pragma unroll
  for (int off = 16; off >= 1; off >>= 1) {
    s  += __shfl_xor(s, off, 32);
    s2 += __shfl_xor(s2, off, 32);
  }
  float mu = s * (1.f / D_);
  float var = s2 * (1.f / D_) - mu * mu;
  float rs = rsqrtf(var + 1e-5f);
#pragma unroll
  for (int i = 0; i < 8; i++) {
    int c = lane + i * 32;
    float v = (t[i] - mu) * rs * g1[c] + b1[c];
    zz[row * D_ + c] = v;
    zzb[row * D_ + c] = f2bf(v);
  }
}

// ---------------------------------------------------------------------------
// Kernel 4: ff = zzb @ Wffb + bff; y = LayerNorm(zz + leaky(ff))
// 128 threads = 4 waves; 16 rows/block; async A-tile stage + Wff prefetch
// ---------------------------------------------------------------------------
__global__ __launch_bounds__(128) void ffln_kernel(
    const unsigned short* __restrict__ zzb, const unsigned short* __restrict__ Wffb,
    const float* __restrict__ zz, const float* __restrict__ bff,
    const float* __restrict__ g2, const float* __restrict__ b2,
    float* __restrict__ y) {
  __shared__ unsigned short a_s[16][DK_];       // unpadded: 8B-aligned rows
  __shared__ unsigned short bT_s[D_][DK_ + 2];  // [n][k] pairs for B operand
  __shared__ float ff_s[16][D_];

  const int tid = threadIdx.x;
  const int wave = tid >> 5, lane = tid & 31;
  const int ln = lane & 15, lh = (lane >> 4) & 1;
  const size_t row0 = (size_t)blockIdx.x * 16;

  f32x8 acc[4];
#pragma unroll
  for (int nt = 0; nt < 4; nt++) acc[nt] = f32x8{};

  for (int k0g = 0; k0g < D_; k0g += 32) {
    __syncthreads();
    {  // async A tile: 16 rows x 64B = 128 x 8B chunks (1 per thread, G=1/wave)
      int r = tid >> 3, pos = (tid & 7) * 4;
      async_b64(lds_off(&a_s[r][pos]),
                (unsigned long long)(zzb + (row0 + r) * D_ + k0g + pos));
    }
    if (k0g + 32 < D_)  // prefetch next Wff k-chunk into caches
      __builtin_prefetch(Wffb + (size_t)(k0g + 32) * D_ + tid * 2, 0, 1);
#pragma unroll
    for (int j = 0; j < 32; j++) {  // B tile: 32 k x 256 n, transposed
      int u = tid + j * 128;
      int kk = u >> 7, np = (u & 127) * 2;
      unsigned int w =
          *(const unsigned int*)(Wffb + (size_t)(k0g + kk) * D_ + np);
      bT_s[np][kk]     = (unsigned short)(w & 0xFFFFu);
      bT_s[np + 1][kk] = (unsigned short)(w >> 16);
    }
    asm volatile("s_wait_asynccnt 0x0" ::: "memory");
    __syncthreads();

    ABu aA;
#pragma unroll
    for (int i = 0; i < 8; i++)
      aA.u[i] = *(const unsigned int*)&a_s[ln][a_k0(i, lh)];
#pragma unroll
    for (int nt = 0; nt < 4; nt++) {
      int ng = wave * 64 + nt * 16;
      ABu bB;
#pragma unroll
      for (int j = 0; j < 8; j++)
        bB.u[j] = *(const unsigned int*)&bT_s[ng + ln][a_k0(j, lh)];
      acc[nt] = wmma_bf16(aA, bB, acc[nt]);
    }
  }

  // bias + LeakyReLU -> LDS
#pragma unroll
  for (int nt = 0; nt < 4; nt++) {
#pragma unroll
    for (int r = 0; r < 8; r++) {
      int m = r + lh * 8;
      int c = wave * 64 + nt * 16 + ln;
      float v = acc[nt][r] + bff[c];
      ff_s[m][c] = (v >= 0.f) ? v : 0.01f * v;
    }
  }
  __syncthreads();

  // LayerNorm over (zz + ff), 8 lanes per row
  {
    int r = tid >> 3, sub = tid & 7;
    const float* zr = zz + (row0 + r) * D_;
    float t[32], s = 0.f, s2 = 0.f;
#pragma unroll
    for (int i = 0; i < 32; i++) {
      int c = sub + i * 8;
      t[i] = zr[c] + ff_s[r][c];
      s += t[i]; s2 += t[i] * t[i];
    }
#pragma unroll
    for (int off = 4; off >= 1; off >>= 1) {
      s  += __shfl_xor(s, off, 32);
      s2 += __shfl_xor(s2, off, 32);
    }
    float mu = s * (1.f / D_);
    float var = s2 * (1.f / D_) - mu * mu;
    float rs = rsqrtf(var + 1e-5f);
#pragma unroll
    for (int i = 0; i < 32; i++) {
      int c = sub + i * 8;
      y[(row0 + r) * D_ + c] = (t[i] - mu) * rs * g2[c] + b2[c];
    }
  }
}

// ---------------------------------------------------------------------------
extern "C" void kernel_launch(void* const* d_in, const int* in_sizes, int n_in,
                              void* d_out, int out_size, void* d_ws,
                              size_t ws_size, hipStream_t stream) {
  const float*         x    = (const float*)d_in[0];
  const unsigned char* mask = (const unsigned char*)d_in[1];  // jnp bool = 1B
  const float*         Wq   = (const float*)d_in[2];
  const float*         Wk   = (const float*)d_in[3];
  const float*         Wv   = (const float*)d_in[4];
  const float*         Wff  = (const float*)d_in[5];
  const float*         bff  = (const float*)d_in[6];
  const float*         g1   = (const float*)d_in[7];
  const float*         b1   = (const float*)d_in[8];
  const float*         g2   = (const float*)d_in[9];
  const float*         b2   = (const float*)d_in[10];
  float* y = (float*)d_out;

  char* ws = (char*)d_ws;
  size_t off = 0;
  auto carve = [&](size_t bytes) -> void* {
    void* p = ws + off;
    off += (bytes + 255) & ~(size_t)255;
    return p;
  };
  unsigned short* qb   = (unsigned short*)carve((size_t)ROWS * DK_ * 2);
  unsigned short* kb   = (unsigned short*)carve((size_t)ROWS * DK_ * 2);
  unsigned short* vbT  = (unsigned short*)carve((size_t)ROWS * D_ * 2);  // [B][D][S]
  float*          sa   = (float*)carve((size_t)ROWS * D_ * 4);
  float*          zz   = (float*)carve((size_t)ROWS * D_ * 4);
  unsigned short* zzb  = (unsigned short*)carve((size_t)ROWS * D_ * 2);
  unsigned short* Wffb = (unsigned short*)carve((size_t)D_ * D_ * 2);
  (void)in_sizes; (void)n_in; (void)out_size; (void)ws_size;

  wconv_kernel<<<(D_ * D_) / 256, 256, 0, stream>>>(Wff, Wffb, D_ * D_);
  qkv_kernel<<<ROWS / 16, 256, 0, stream>>>(x, Wq, Wk, Wv, qb, kb, vbT);
  attn_kernel<<<dim3(S_ / 64, B_), 256, 0, stream>>>(qb, kb, vbT, mask, sa);
  ln1_kernel<<<ROWS / 8, 256, 0, stream>>>(x, sa, g1, b1, zz, zzb);
  ffln_kernel<<<ROWS / 16, 128, 0, stream>>>(zzb, Wffb, zz, bff, g2, b2, y);
}
